// GlobalPointer_5884105196193
// MI455X (gfx1250) — matrix operand order repaired and checked by reading, not compiled, not run on hardware
//
#include <hip/hip_runtime.h>

// Problem dims (fixed by the reference)
#define HEADS     12
#define HEAD_SIZE 64
#define BS        16
#define SEQ       512
#define HID       768
#define NCOL      (HEADS * 2 * HEAD_SIZE)   // 1536
#define NEG_BIG   1000000000000.0f

typedef __attribute__((ext_vector_type(16))) _Float16 v16h;
typedef __attribute__((ext_vector_type(8)))  _Float16 v8h;
typedef __attribute__((ext_vector_type(8)))  float    v8f;

#define WMMA_F16(A, B, C) \
  __builtin_amdgcn_wmma_f32_16x16x32_f16(false, (A), false, (B), (short)0, (C), false, false)

// ---------------------------------------------------------------------------
// WMMA fragment loaders (CDNA5 ISA 7.12.2 layouts, wave32)
//
// A 16x32 f16 (MxK): lane L holds row M = L&15; K chunks of 8 at
//   k0 + 8*(L>>4) and k0 + 8*(L>>4) + 16.
// B 32x16 f16 (KxN): lane L holds column N = L&15; K chunks of 8 at
//   k0 + 8*(L>>4) and k0 + 8*(L>>4) + 16 (same as A).  We store B transposed
//   ([N,K] row-major) so each chunk is a 16-byte contiguous load.
// C/D f32 16x16: VGPR r, lane L -> M = r + 8*(L>>4), N = L&15.
// ---------------------------------------------------------------------------
__device__ __forceinline__ v16h load_a_f16(const _Float16* __restrict__ base,
                                           int ld, int row0, int k0, int lane) {
  int row = row0 + (lane & 15);
  int kk  = k0 + ((lane >> 4) << 3);
  const _Float16* p = base + (size_t)row * ld + kk;
  v8h lo = *(const v8h*)p;          // K = kk .. kk+7   (16B aligned)
  v8h hi = *(const v8h*)(p + 16);   // K = kk+16 .. kk+23
  v16h r;
#pragma unroll
  for (int i = 0; i < 8; ++i) { r[i] = lo[i]; r[i + 8] = hi[i]; }
  return r;
}

// Same A layout but source is f32 (hidden): convert while loading, so hidden
// is read exactly once per wave and never materialized in f16.
__device__ __forceinline__ v16h load_a_f32cvt(const float* __restrict__ base,
                                              int ld, int row0, int k0, int lane) {
  int row = row0 + (lane & 15);
  int kk  = k0 + ((lane >> 4) << 3);
  const float* p = base + (size_t)row * ld + kk;
  v8f lo = *(const v8f*)p;          // 32B aligned
  v8f hi = *(const v8f*)(p + 16);
  v16h r;
#pragma unroll
  for (int i = 0; i < 8; ++i) { r[i] = (_Float16)lo[i]; r[i + 8] = (_Float16)hi[i]; }
  return r;
}

// baseT is [N, K] row-major (i.e. B^T), ld = K stride
__device__ __forceinline__ v16h load_b_f16(const _Float16* __restrict__ baseT,
                                           int ld, int col0, int k0, int lane) {
  int col = col0 + (lane & 15);
  int kk  = k0 + ((lane >> 4) << 3);
  const _Float16* p = baseT + (size_t)col * ld + kk;
  v8h lo = *(const v8h*)p;
  v8h hi = *(const v8h*)(p + 16);
  v16h r;
#pragma unroll
  for (int i = 0; i < 8; ++i) { r[i] = lo[i]; r[i + 8] = hi[i]; }
  return r;
}

// ---------------------------------------------------------------------------
// Kernel 1: W [768,1536] f32 row-major -> Wt [1536,768] f16 row-major.
// LDS tile transpose: both the global read and write are coalesced.
// ---------------------------------------------------------------------------
__global__ void wt_convert(const float* __restrict__ W, _Float16* __restrict__ Wt) {
  __shared__ float tile[32][33];  // +1 pad: avoid 32-way LDS bank conflicts
  int n0 = blockIdx.x * 32;       // column block of W
  int k0 = blockIdx.y * 32;       // row block of W
  int tx = threadIdx.x, ty = threadIdx.y;
  tile[ty][tx] = W[(size_t)(k0 + ty) * NCOL + (n0 + tx)];
  __syncthreads();
  Wt[(size_t)(n0 + ty) * HID + (k0 + tx)] = (_Float16)tile[tx][ty];
}

// ---------------------------------------------------------------------------
// Kernel 2: projection GEMM + bias + interleaved RoPE.
// Block = 128 threads (4 waves). Each wave computes a 16(M) x 32(N) tile so
// one A fragment (and its f32->f16 converts) feeds TWO WMMAs. Block covers
// rows m0..m0+15, cols n0..n0+127.  16-row tiles never straddle a batch
// (512 % 16 == 0); each 16-col tile sits inside one q/k 64-half (wave N base
// is a multiple of 32).  Results are rotated (RoPE) and stored as f16 into
// Q/K with layout [b, h, seq, 64] — exactly what the attention fragments eat.
// ---------------------------------------------------------------------------
__global__ void proj_rope(const float* __restrict__ hidden,
                          const _Float16* __restrict__ Wt,
                          const float* __restrict__ bias,
                          _Float16* __restrict__ Qf,
                          _Float16* __restrict__ Kf) {
  int lane = threadIdx.x & 31;
  int wave = threadIdx.x >> 5;
  int m0  = blockIdx.x * 16;
  int n0w = blockIdx.y * 128 + wave * 32;   // wave owns cols n0w .. n0w+31

  v8f acc0 = {}, acc1 = {};
#pragma unroll 4
  for (int k0 = 0; k0 < HID; k0 += 32) {
    v16h a  = load_a_f32cvt(hidden, HID, m0, k0, lane);
    v16h b0 = load_b_f16(Wt, HID, n0w,      k0, lane);
    v16h b1 = load_b_f16(Wt, HID, n0w + 16, k0, lane);
    if (k0 + 32 < HID)   // pull the next B strip toward the WGP
      __builtin_prefetch(Wt + (size_t)(n0w + (lane & 15)) * HID + k0 + 32, 0, 1);
    acc0 = WMMA_F16(a, b0, acc0);
    acc1 = WMMA_F16(a, b1, acc1);
  }

  int hi8 = (lane >> 4) << 3;
#pragma unroll
  for (int t = 0; t < 2; ++t) {
    v8f acc = t ? acc1 : acc0;
    int n    = n0w + t * 16 + (lane & 15);  // projection column
    int head = n >> 7;                      // n / 128
    int qk   = (n >> 6) & 1;                // 0 = q, 1 = k (constant per tile)
    int d    = n & 63;                      // head dim
    float bn   = bias[n];
    // inv_freq = 10000^(-2*(d/2)/64); 2*(d>>1) == d & ~1
    float invf = powf(10000.0f, -(float)(d & ~1) / (float)HEAD_SIZE);
    _Float16* dst = qk ? Kf : Qf;
#pragma unroll
    for (int r = 0; r < 8; ++r) {
      int m     = m0 + r + hi8;     // global row = batch*512 + pos
      int batch = m >> 9;
      int pos   = m & (SEQ - 1);
      float x = acc[r] + bn;
      // Interleaved rotate-half partner lives in the neighboring lane
      // (d pairs are lane pairs; tile base d is even so xor-1 never crosses
      // the lane-16 row boundary).
      float partner = __shfl_xor(x, 1, 32);
      float rot = (d & 1) ? partner : -partner;
      float sn, cs;
      sincosf((float)pos * invf, &sn, &cs);
      float val = x * cs + rot * sn;
      dst[(((size_t)(batch * HEADS + head)) * SEQ + pos) * HEAD_SIZE + d] =
          (_Float16)val;
    }
  }
}

// ---------------------------------------------------------------------------
// Kernel 3: attention scores + masks + scale.
// Grid (SEQ/64, SEQ/128, BS*HEADS); block = 128 threads (4 waves).
// Each wave computes a 32(M) x 64(N) strip: two A fragments share every B
// fragment, so K = 64 -> 2 k-steps x 4 n-tiles x 2 m-tiles = 16 WMMAs per
// wave with only 8 B loads.  Masks are applied BEFORE the 1/sqrt(64) scale,
// matching the reference's `(logits - masks*NEG_BIG) / 8`.
// ---------------------------------------------------------------------------
__global__ void attn_scores(const _Float16* __restrict__ Qf,
                            const _Float16* __restrict__ Kf,
                            const int* __restrict__ amask,
                            float* __restrict__ out) {
  int lane = threadIdx.x & 31;
  int wave = threadIdx.x >> 5;
  int bz    = blockIdx.z;          // b*HEADS + h
  int batch = bz / HEADS;
  int n0  = blockIdx.x * 64;
  int m0w = blockIdx.y * 128 + wave * 32;   // wave owns rows m0w .. m0w+31

  const _Float16* Qh = Qf + (size_t)bz * SEQ * HEAD_SIZE;
  const _Float16* Kh = Kf + (size_t)bz * SEQ * HEAD_SIZE;

  v8f acc[2][4] = {};
#pragma unroll
  for (int k0 = 0; k0 < HEAD_SIZE; k0 += 32) {
    v16h a0 = load_a_f16(Qh, HEAD_SIZE, m0w,      k0, lane);
    v16h a1 = load_a_f16(Qh, HEAD_SIZE, m0w + 16, k0, lane);
#pragma unroll
    for (int j = 0; j < 4; ++j) {
      // B = K^T: column n of K^T is row n of K (contiguous in [seq,64]).
      v16h b = load_b_f16(Kh, HEAD_SIZE, n0 + j * 16, k0, lane);
      acc[0][j] = WMMA_F16(a0, b, acc[0][j]);
      acc[1][j] = WMMA_F16(a1, b, acc[1][j]);
    }
  }

  const int* am = amask + batch * SEQ;
  int hi8 = (lane >> 4) << 3;
  float am_m[2][8];
#pragma unroll
  for (int i = 0; i < 2; ++i)
#pragma unroll
    for (int r = 0; r < 8; ++r)
      am_m[i][r] = (float)am[m0w + i * 16 + r + hi8];

  const float scale = 0.125f;  // 1/sqrt(HEAD_SIZE)
#pragma unroll
  for (int j = 0; j < 4; ++j) {
    int n = n0 + j * 16 + (lane & 15);
    float am_n = (float)am[n];
#pragma unroll
    for (int i = 0; i < 2; ++i) {
#pragma unroll
      for (int r = 0; r < 8; ++r) {
        int m = m0w + i * 16 + r + hi8;
        float v = acc[i][j][r];
        v -= (1.0f - am_m[i][r] * am_n) * NEG_BIG;   // pad mask
        if (n < m) v -= NEG_BIG;                     // tril(k=-1) mask
        out[((size_t)bz * SEQ + m) * SEQ + n] = v * scale;
      }
    }
  }
}

// ---------------------------------------------------------------------------
extern "C" void kernel_launch(void* const* d_in, const int* in_sizes, int n_in,
                              void* d_out, int out_size, void* d_ws, size_t ws_size,
                              hipStream_t stream) {
  (void)in_sizes; (void)n_in; (void)out_size; (void)ws_size;

  const float* hidden = (const float*)d_in[0];  // [16,512,768] f32
  const int*   amask  = (const int*)d_in[1];    // [16,512] i32
  const float* W      = (const float*)d_in[2];  // [768,1536] f32
  const float* bias   = (const float*)d_in[3];  // [1536] f32
  float* out = (float*)d_out;                   // [16,12,512,512] f32

  // Workspace carve-out (256B-aligned offsets for vector loads):
  //   Wt : 1536*768  f16 =  2,359,296 B
  //   Qf : 16*12*512*64 f16 = 12,582,912 B
  //   Kf : same                      -> total ~27.5 MB
  char* ws = (char*)d_ws;
  _Float16* Wt = (_Float16*)(ws);
  _Float16* Qf = (_Float16*)(ws + 2359296u);
  _Float16* Kf = (_Float16*)(ws + 2359296u + 12582912u);

  wt_convert<<<dim3(NCOL / 32, HID / 32), dim3(32, 32), 0, stream>>>(W, Wt);

  proj_rope<<<dim3((BS * SEQ) / 16, NCOL / 128), 128, 0, stream>>>(
      hidden, Wt, bias, Qf, Kf);

  attn_scores<<<dim3(SEQ / 64, SEQ / 128, BS * HEADS), 128, 0, stream>>>(
      Qf, Kf, amask, out);
}
